// Meta_67078799229377
// MI455X (gfx1250) — matrix operand order repaired and checked by reading, not compiled, not run on hardware
//
#include <hip/hip_runtime.h>
#include <hip/hip_bf16.h>
#include <math.h>

typedef __attribute__((ext_vector_type(16))) _Float16 v16h;
typedef __attribute__((ext_vector_type(8)))  _Float16 h8;
typedef __attribute__((ext_vector_type(8)))  float    v8f;

// ---------------------------------------------------------------------------
// conv1: 3->32 ch, 64x64 -> 32x32, stride 2, SAME (pad_lo=0), ReLU.
// Reads support_x / query_x directly (logical concat on axis 1).
// ---------------------------------------------------------------------------
__global__ void conv1_kernel(const float* __restrict__ sup,
                             const float* __restrict__ qry,
                             const float* __restrict__ wt,
                             const float* __restrict__ bias,
                             float* __restrict__ out) {
  int n  = blockIdx.x >> 5;   // 24 images
  int co = blockIdx.x & 31;   // 32 out channels
  int b = n / 6, s = n % 6;
  const float* src = (s < 5) ? (sup + (size_t)(b * 5 + s) * 3 * 64 * 64)
                             : (qry + (size_t)b * 3 * 64 * 64);
  float bv = bias[co];
  for (int px = threadIdx.x; px < 32 * 32; px += blockDim.x) {
    int oy = px >> 5, ox = px & 31;
    float acc = bv;
    for (int ci = 0; ci < 3; ++ci) {
      const float* ip = src + (size_t)ci * 64 * 64;
      const float* wp = wt + (size_t)(co * 3 + ci) * 9;
      for (int ky = 0; ky < 3; ++ky) {
        int iy = oy * 2 + ky;
        if (iy >= 64) continue;
        for (int kx = 0; kx < 3; ++kx) {
          int ix = ox * 2 + kx;
          if (ix >= 64) continue;
          acc += ip[iy * 64 + ix] * wp[ky * 3 + kx];
        }
      }
    }
    out[(size_t)(n * 32 + co) * 1024 + px] = fmaxf(acc, 0.f);
  }
}

// ---------------------------------------------------------------------------
// Generic stride-2 3x3 SAME conv + ReLU (layers 2 and 3).
// ---------------------------------------------------------------------------
__global__ void conv_s2_kernel(const float* __restrict__ in,
                               const float* __restrict__ wt,
                               const float* __restrict__ bias,
                               float* __restrict__ out,
                               int Cin, int Hin, int Cout, int Hout) {
  int n  = blockIdx.x / Cout;
  int co = blockIdx.x % Cout;
  float bv = bias[co];
  int HW = Hout * Hout;
  for (int px = threadIdx.x; px < HW; px += blockDim.x) {
    int oy = px / Hout, ox = px % Hout;
    float acc = bv;
    for (int ci = 0; ci < Cin; ++ci) {
      const float* ip = in + (size_t)(n * Cin + ci) * Hin * Hin;
      const float* wp = wt + (size_t)(co * Cin + ci) * 9;
      for (int ky = 0; ky < 3; ++ky) {
        int iy = oy * 2 + ky;
        if (iy >= Hin) continue;
        for (int kx = 0; kx < 3; ++kx) {
          int ix = ox * 2 + kx;
          if (ix >= Hin) continue;
          acc += ip[iy * Hin + ix] * wp[ky * 3 + kx];
        }
      }
    }
    out[(size_t)(n * Cout + co) * HW + px] = fmaxf(acc, 0.f);
  }
}

// ---------------------------------------------------------------------------
// Spatial mean pool (64 ch x 8x8) + logits = pooled @ Wlog + blog.
// ---------------------------------------------------------------------------
__global__ void pool_logits_kernel(const float* __restrict__ feat3,
                                   const float* __restrict__ Wlog,
                                   const float* __restrict__ blog,
                                   float* __restrict__ logits) {
  __shared__ float pooled[64];
  int n = blockIdx.x, t = threadIdx.x;
  const float* fp = feat3 + (size_t)n * 4096 + t * 64;
  float sum = 0.f;
  for (int p = 0; p < 64; ++p) sum += fp[p];
  pooled[t] = sum * (1.f / 64.f);
  __syncthreads();
  float acc = blog[t];
  for (int c = 0; c < 64; ++c) acc += pooled[c] * Wlog[c * 64 + t];
  logits[n * 64 + t] = acc;
}

// ---------------------------------------------------------------------------
// u = a @ Wg1[:66]; v = a @ Wg1[66:] + bg1, a = [feat3 | coords].
// Written directly in fp16 so the relation kernel's A-build is packed-half.
// ---------------------------------------------------------------------------
__global__ void uv_kernel(const float* __restrict__ feat3,
                          const float* __restrict__ Wg1,
                          const float* __restrict__ bg1,
                          _Float16* __restrict__ u, _Float16* __restrict__ v) {
  int row = blockIdx.x;          // 1536 = n*64 + p
  int n = row >> 6, p = row & 63;
  int k = threadIdx.x;           // 128
  const float* fp = feat3 + (size_t)n * 4096 + p;  // channel stride 64
  float su = 0.f, sv = 0.f;
  for (int c = 0; c < 64; ++c) {
    float a = fp[c * 64];
    su += a * Wg1[c * 128 + k];
    sv += a * Wg1[(66 + c) * 128 + k];
  }
  float cy = (float)(p >> 3) * 0.125f;
  float cx = (float)(p & 7) * 0.125f;
  su += cy * Wg1[64 * 128 + k] + cx * Wg1[65 * 128 + k];
  sv += cy * Wg1[130 * 128 + k] + cx * Wg1[131 * 128 + k] + bg1[k];
  u[(size_t)row * 128 + k] = (_Float16)su;
  v[(size_t)row * 128 + k] = (_Float16)sv;
}

// ---------------------------------------------------------------------------
// float -> fp16 conversion (Wg2).
// ---------------------------------------------------------------------------
__global__ void cvt_half_kernel(const float* __restrict__ src,
                                _Float16* __restrict__ dst, int nElem) {
  int i = blockIdx.x * blockDim.x + threadIdx.x;
  if (i < nElem) dst[i] = (_Float16)src[i];
}

// ---------------------------------------------------------------------------
// Zero-fill (xf accumulator).
// ---------------------------------------------------------------------------
__global__ void zero_kernel(float* __restrict__ p, int nElem) {
  int i = blockIdx.x * blockDim.x + threadIdx.x;
  if (i < nElem) p[i] = 0.f;
}

// ---------------------------------------------------------------------------
// Relation kernel (the 9.7-GFLOP GEMM), fused:
//   x_f[b,i,j,:] = sum_{p,q} relu( relu(u[b,j,p,:]+v[b,i,q,:]) @ Wg2 + bg2 )
// 144 blocks (one per b,i,j), 4 waves; wave w owns q-tile w:
//   - its v-row fragments are loop-invariant (loaded once),
//   - each A-build (packed-fp16 add+max, b128 loads) feeds 16 WMMAs.
// Wg2 fragments (4 N-tiles x 4 K-frags) are register-resident.
// ---------------------------------------------------------------------------
__global__ __launch_bounds__(128) void relation_kernel(
    const _Float16* __restrict__ uh, const _Float16* __restrict__ vh,
    const _Float16* __restrict__ Bh, const float* __restrict__ bg2,
    float* __restrict__ xf) {
  int blk = blockIdx.x;                 // b*36 + i*6 + j
  int bb = blk / 36;
  int rem = blk % 36;
  int i = rem / 6, j = rem % 6;
  int wave  = threadIdx.x >> 5;         // q-tile owned by this wave (0..3)
  int lane  = threadIdx.x & 31;
  int mrow  = lane & 15;
  int hsel  = lane >> 4;

  const _Float16* ubase = uh + (size_t)((bb * 6 + j) * 64) * 128;
  const _Float16* vbase = vh + (size_t)((bb * 6 + i) * 64) * 128;

  // Wg2 B fragments: element e -> K = kf*32 + (e>>3)*16 + hsel*8 + (e&7), N = nt*16 + mrow
  v16h Bfrag[4][4];
#pragma unroll
  for (int nt = 0; nt < 4; ++nt)
#pragma unroll
    for (int kf = 0; kf < 4; ++kf)
#pragma unroll
      for (int e = 0; e < 16; ++e)
        Bfrag[nt][kf][e] = Bh[(kf * 32 + (e >> 3) * 16 + hsel * 8 + (e & 7)) * 64 + nt * 16 + mrow];

  float bias[4];
#pragma unroll
  for (int nt = 0; nt < 4; ++nt) bias[nt] = bg2[nt * 16 + mrow];

  // This wave's A-row q = wave*16 + mrow is fixed: preload its v chunks.
  // A element e<8  -> K = kf*32      + hsel*8 + e      (16B chunk kf*4+hsel)
  // A element e>=8 -> K = kf*32 + 16 + hsel*8 + (e-8)  (16B chunk kf*4+2+hsel)
  const h8* vq8 = (const h8*)(vbase + (size_t)(wave * 16 + mrow) * 128);
  h8 vchunk[8];
#pragma unroll
  for (int c = 0; c < 8; ++c) {
    int kf = c >> 1, grp = c & 1;
    vchunk[c] = vq8[kf * 4 + grp * 2 + hsel];
  }

  float psum[4] = {0.f, 0.f, 0.f, 0.f};
  for (int p = 0; p < 64; ++p) {
    const h8* up8 = (const h8*)(ubase + (size_t)p * 128);
    v16h A[4];
#pragma unroll
    for (int kf = 0; kf < 4; ++kf) {
      h8 z = {};
      h8 lo = up8[kf * 4 + hsel]     + vchunk[kf * 2 + 0];
      h8 hi = up8[kf * 4 + 2 + hsel] + vchunk[kf * 2 + 1];
      lo = __builtin_elementwise_max(lo, z);
      hi = __builtin_elementwise_max(hi, z);
      A[kf] = __builtin_shufflevector(lo, hi, 0, 1, 2, 3, 4, 5, 6, 7,
                                      8, 9, 10, 11, 12, 13, 14, 15);
    }
#pragma unroll
    for (int nt = 0; nt < 4; ++nt) {
      v8f C = {};
#pragma unroll
      for (int kf = 0; kf < 4; ++kf)
        C = __builtin_amdgcn_wmma_f32_16x16x32_f16(false, A[kf], false,
                                                   Bfrag[nt][kf], (short)0, C,
                                                   false, false);
#pragma unroll
      for (int r = 0; r < 8; ++r) {
        float g = C[r] + bias[nt];
        psum[nt] += (g > 0.f ? g : 0.f);
      }
    }
  }
  // lanes L and L^16 hold complementary row halves of the same column;
  // waves hold disjoint q-tiles -> reduce across waves with f32 atomics.
#pragma unroll
  for (int nt = 0; nt < 4; ++nt) {
    float tot = psum[nt] + __shfl_xor(psum[nt], 16, 32);
    if (hsel == 0) atomicAdd(&xf[(size_t)blk * 64 + nt * 16 + mrow], tot);
  }
}

// ---------------------------------------------------------------------------
// score = sigmoid(relu(x_f @ Wf1 + bf1) @ Wf2 + bf2), per (b,i,j).
// ---------------------------------------------------------------------------
__global__ void score_kernel(const float* __restrict__ xf,
                             const float* __restrict__ Wf1,
                             const float* __restrict__ bf1,
                             const float* __restrict__ Wf2,
                             const float* __restrict__ bf2,
                             float* __restrict__ score) {
  __shared__ float h1[16];
  int idx = blockIdx.x;      // 144
  int t = threadIdx.x;       // 32 threads, 16 active for hidden layer
  if (t < 16) {
    float acc = bf1[t];
    const float* xp = xf + (size_t)idx * 64;
    for (int c = 0; c < 64; ++c) acc += xp[c] * Wf1[c * 16 + t];
    h1[t] = acc > 0.f ? acc : 0.f;
  }
  __syncthreads();
  if (t == 0) {
    float acc = bf2[0];
    for (int c = 0; c < 16; ++c) acc += h1[c] * Wf2[c];
    score[idx] = 1.f / (1.f + expf(-acc));
  }
}

// ---------------------------------------------------------------------------
// Final scalar losses: cls_loss, rn_loss, sym_loss -> d_out.
// ---------------------------------------------------------------------------
__global__ void loss_kernel(const float* __restrict__ logits,
                            const float* __restrict__ score,
                            const int* __restrict__ sy,
                            const int* __restrict__ qy,
                            float* __restrict__ out) {
  if (threadIdx.x != 0 || blockIdx.x != 0) return;
  float cls = 0.f;
  for (int n = 0; n < 24; ++n) {
    int b = n / 6, s = n % 6;
    int lab = (s < 5) ? sy[b * 5 + s] : qy[b];
    const float* lg = logits + n * 64;
    float mx = lg[0];
    for (int k = 1; k < 64; ++k) mx = fmaxf(mx, lg[k]);
    float se = 0.f;
    for (int k = 0; k < 64; ++k) se += expf(lg[k] - mx);
    cls += mx + logf(se) - lg[lab];
  }
  cls *= (1.f / 24.f);

  float euc = 0.f, symtot = 0.f;
  for (int b = 0; b < 4; ++b) {
    int lab6[6];
    for (int s = 0; s < 5; ++s) lab6[s] = sy[b * 5 + s];
    lab6[5] = qy[b];
    float sn = 0.f, an = 0.f;
    for (int i = 0; i < 6; ++i)
      for (int jj = 0; jj < 6; ++jj) {
        float P  = score[b * 36 + i * 6 + jj];
        float Pt = score[b * 36 + jj * 6 + i];
        float symv = 0.5f * (P + Pt), antiv = 0.5f * (P - Pt);
        sn += symv * symv;
        an += antiv * antiv;
        float y = (lab6[i] == lab6[jj]) ? 1.f : 0.f;
        float d = P - y;
        euc += d * d;
      }
    sn = sqrtf(sn);
    an = sqrtf(an);
    symtot += (sn - an) / (sn + an);
  }
  float sym_loss = symtot * 0.25f;
  euc *= (1.f / 144.f);
  out[0] = cls;
  out[1] = euc - 0.1f * sym_loss;
  out[2] = sym_loss;
}

// ---------------------------------------------------------------------------
extern "C" void kernel_launch(void* const* d_in, const int* in_sizes, int n_in,
                              void* d_out, int out_size, void* d_ws, size_t ws_size,
                              hipStream_t stream) {
  const float* support_x = (const float*)d_in[0];
  const int*   support_y = (const int*)d_in[1];
  const float* query_x   = (const float*)d_in[2];
  const int*   query_y   = (const int*)d_in[3];
  const float* k1   = (const float*)d_in[4];
  const float* bc1  = (const float*)d_in[5];
  const float* k2   = (const float*)d_in[6];
  const float* bc2  = (const float*)d_in[7];
  const float* k3   = (const float*)d_in[8];
  const float* bc3  = (const float*)d_in[9];
  const float* Wlog = (const float*)d_in[10];
  const float* blog = (const float*)d_in[11];
  const float* Wg1  = (const float*)d_in[12];
  const float* bg1  = (const float*)d_in[13];
  const float* Wg2  = (const float*)d_in[14];
  const float* bg2  = (const float*)d_in[15];
  const float* Wf1  = (const float*)d_in[16];
  const float* bf1  = (const float*)d_in[17];
  const float* Wf2  = (const float*)d_in[18];
  const float* bf2  = (const float*)d_in[19];
  float* out = (float*)d_out;

  // Workspace carve (float offsets)
  float* ws = (float*)d_ws;
  const size_t OFF_F1 = 0;                     // 24*32*32*32   = 786432
  const size_t OFF_F2 = OFF_F1 + 786432;       // 24*48*16*16   = 294912
  const size_t OFF_F3 = OFF_F2 + 294912;       // 24*64*8*8     = 98304
  const size_t OFF_LG = OFF_F3 + 98304;        // 24*64         = 1536
  const size_t OFF_UH = OFF_LG + 1536;         // 196608 halves = 98304 floats
  const size_t OFF_VH = OFF_UH + 98304;        // 196608 halves = 98304 floats
  const size_t OFF_XF = OFF_VH + 98304;        // 4*36*64       = 9216
  const size_t OFF_SC = OFF_XF + 9216;         // 144
  const size_t OFF_WH = OFF_SC + 144;          // 8192 halves (4096 float slots)
  float* feat1  = ws + OFF_F1;
  float* feat2  = ws + OFF_F2;
  float* feat3  = ws + OFF_F3;
  float* logits = ws + OFF_LG;
  _Float16* uh  = (_Float16*)(ws + OFF_UH);
  _Float16* vh  = (_Float16*)(ws + OFF_VH);
  float* xf     = ws + OFF_XF;
  float* score  = ws + OFF_SC;
  _Float16* wg2h = (_Float16*)(ws + OFF_WH);

  // Conv tower
  conv1_kernel<<<dim3(24 * 32), dim3(256), 0, stream>>>(support_x, query_x, k1, bc1, feat1);
  conv_s2_kernel<<<dim3(24 * 48), dim3(256), 0, stream>>>(feat1, k2, bc2, feat2, 32, 32, 48, 16);
  conv_s2_kernel<<<dim3(24 * 64), dim3(64), 0, stream>>>(feat2, k3, bc3, feat3, 48, 16, 64, 8);

  // Classifier head
  pool_logits_kernel<<<dim3(24), dim3(64), 0, stream>>>(feat3, Wlog, blog, logits);

  // Relation-net projections (fp16 out) + weight convert + xf zero
  uv_kernel<<<dim3(1536), dim3(128), 0, stream>>>(feat3, Wg1, bg1, uh, vh);
  cvt_half_kernel<<<dim3(32), dim3(256), 0, stream>>>(Wg2, wg2h, 128 * 64);
  zero_kernel<<<dim3(36), dim3(256), 0, stream>>>(xf, 9216);

  // Fused pairwise relation GEMM (WMMA)
  relation_kernel<<<dim3(144), dim3(128), 0, stream>>>(uh, vh, wg2h, bg2, xf);

  // Score MLP + losses
  score_kernel<<<dim3(144), dim3(32), 0, stream>>>(xf, Wf1, bf1, Wf2, bf2, score);
  loss_kernel<<<dim3(1), dim3(32), 0, stream>>>(logits, score, support_y, query_y, out);
}
